// GIN_18038862643738
// MI455X (gfx1250) — matrix operand order repaired and checked by reading, not compiled, not run on hardware
//
#include <hip/hip_runtime.h>
#include <cstdint>
#include <cstddef>

// ---------------------------------------------------------------------------
// Types for CDNA5 WMMA (wave32): v16bf A/B fragments, v8f accumulator.
// ---------------------------------------------------------------------------
typedef __attribute__((ext_vector_type(16))) __bf16        v16bf;
typedef __attribute__((ext_vector_type(8)))  float         v8f;
typedef __attribute__((ext_vector_type(4)))  unsigned int  u32x4;

struct Frag {
    union { v16bf v; u32x4 q[2]; };
};

__device__ __forceinline__ unsigned short f32_to_bf16_rne(float f) {
    unsigned int u = __float_as_uint(f);
    u += 0x7fffu + ((u >> 16) & 1u);           // round-to-nearest-even
    return (unsigned short)(u >> 16);
}

// ---------------------------------------------------------------------------
// 1) Zero the aggregation buffer (workspace is poisoned, must init ourselves).
// ---------------------------------------------------------------------------
__global__ void gin_zero_f32(float* __restrict__ p, int n) {
    int i = blockIdx.x * blockDim.x + threadIdx.x;
    if (i < n) p[i] = 0.0f;
}

// ---------------------------------------------------------------------------
// 2) Edge scatter: agg[dst[e]] += h[src[e]], float4 per thread, hardware
//    global_atomic_add_f32.  agg (<=102MB) fits in the 192MB L2, so the
//    random-address atomics and row gathers mostly resolve in L2.
//    groupShift = log2(C/4); total = E << groupShift.
// ---------------------------------------------------------------------------
__global__ void gin_scatter_add(const float* __restrict__ h,
                                const int*   __restrict__ src,
                                const int*   __restrict__ dst,
                                float*       __restrict__ agg,
                                int C, int groupShift, int total) {
    int i = blockIdx.x * blockDim.x + threadIdx.x;
    if (i >= total) return;
    int e = i >> groupShift;
    int g = i & ((1 << groupShift) - 1);
    int s = src[e];
    int d = dst[e];
    const float4 v = reinterpret_cast<const float4*>(h + (size_t)s * C)[g];
    float* o = agg + (size_t)d * C + (size_t)g * 4;
    unsafeAtomicAdd(o + 0, v.x);
    unsafeAtomicAdd(o + 1, v.y);
    unsafeAtomicAdd(o + 2, v.z);
    unsafeAtomicAdd(o + 3, v.w);
}

// ---------------------------------------------------------------------------
// 3) Combine (1+eps)*h + agg and convert to bf16 (GEMM A operand).
// ---------------------------------------------------------------------------
__global__ void gin_combine_bf16(const float* __restrict__ h,
                                 const float* __restrict__ agg,
                                 const float* __restrict__ epsp,
                                 unsigned short* __restrict__ zbf,
                                 int total) {
    int i = blockIdx.x * blockDim.x + threadIdx.x;
    if (i >= total) return;
    float eps = epsp[0];
    zbf[i] = f32_to_bf16_rne((1.0f + eps) * h[i] + agg[i]);
}

// ---------------------------------------------------------------------------
// 4) Transpose + convert W (K x M, row major) -> WT (M x K, row major) bf16.
// ---------------------------------------------------------------------------
__global__ void gin_wt_bf16(const float* __restrict__ w,
                            unsigned short* __restrict__ wT,
                            int K, int M, int total) {
    int i = blockIdx.x * blockDim.x + threadIdx.x;
    if (i >= total) return;
    int k = i / M;
    int m = i - k * M;
    wT[(size_t)m * K + k] = f32_to_bf16_rne(w[i]);
}

// ---------------------------------------------------------------------------
// 5) GEMM: out[N x M] = A[N x K](bf16) * W[K x M](bf16, stored as WT MxK)
//          + bias, optional ReLU, f32 out.  K/M/RELU are template params so
//    all fragment addresses are a single base register + 24-bit immediate
//    instruction offsets (no pointer arrays -> no spills).
//
//    One wave32 computes a 16x64 output tile (4 adjacent 16x16 WMMA tiles);
//    the A fragment is reused by 4 v_wmma_f32_16x16x32_bf16.  The k-loop is
//    software-pipelined over two disjoint fragment sets (K % 64 == 0) so
//    loads for chunk k+32 are in flight while chunk k is consumed.
//
//    Fragment layouts (ISA 7.12.2, 16-bit):
//      A (16x32): lane holds row lane&15; two contiguous 8-elem chunks at
//                 K offsets (lane>>4)*8 and (lane>>4)*8+16.
//      B (32x16): lane holds column lane&15 == row lane&15 of WT; two
//                 contiguous 8-elem chunks at K offsets (lane>>4)*8 and (lane>>4)*8+16.
//      C/D:       VGPR r -> row r + (lane>>4)*8, col lane&15.
// ---------------------------------------------------------------------------
template <int K, int M, bool RELU>
__global__ void gin_gemm_wmma_bf16(const unsigned short* __restrict__ A,
                                   const unsigned short* __restrict__ WT,
                                   const float* __restrict__ bias,
                                   float* __restrict__ out,
                                   int nGroupsM, int nWaves) {
    constexpr int BSTRIDE = 16 * K;             // elements between adjacent B tiles

    int wave = (blockIdx.x * blockDim.x + threadIdx.x) >> 5;
    if (wave >= nWaves) return;                 // wave-uniform: EXEC stays all-1s
    int tn  = wave / nGroupsM;                  // row-block (16 rows)
    int tmg = wave - tn * nGroupsM;             // column group (64 cols)

    int lane = threadIdx.x & 31;
    int half = lane >> 4;                       // 0: lanes 0-15, 1: lanes 16-31
    int l16  = lane & 15;

    const unsigned short* arow  = A  + (size_t)(tn * 16 + l16) * K + (size_t)half * 8;
    const unsigned short* bbase = WT + (size_t)(tmg * 64 + l16) * K + (size_t)half * 8;

    auto load_a = [&](Frag& a, int k) {
        a.q[0] = *reinterpret_cast<const u32x4*>(arow + k);        // K: ka..ka+7
        a.q[1] = *reinterpret_cast<const u32x4*>(arow + k + 16);   // K: ka+16..ka+23
    };
    auto load_b4 = [&](Frag b[4], int k) {
#pragma unroll
        for (int j = 0; j < 4; ++j) {
            b[j].q[0] = *reinterpret_cast<const u32x4*>(bbase + j * BSTRIDE + k);
            b[j].q[1] = *reinterpret_cast<const u32x4*>(bbase + j * BSTRIDE + k + 16);
        }
    };

    v8f acc[4] = {{}, {}, {}, {}};
    auto mma4 = [&](const Frag& a, const Frag b[4]) {
#pragma unroll
        for (int j = 0; j < 4; ++j) {
            acc[j] = __builtin_amdgcn_wmma_f32_16x16x32_bf16(
                         /*neg_a=*/false, a.v, /*neg_b=*/false, b[j].v,
                         /*c_mod=*/(short)0, acc[j],
                         /*reuse_a=*/false, /*reuse_b=*/false);
        }
    };

    // Software-pipelined ping-pong over 64-deep K super-chunks.
    Frag a0, a1;
    Frag b0[4], b1[4];
    load_a(a0, 0);
    load_b4(b0, 0);
#pragma unroll 1
    for (int k = 0; k < K; k += 64) {
        load_a(a1, k + 32);                     // prefetch odd chunk
        load_b4(b1, k + 32);
        mma4(a0, b0);                           // consume even chunk
        if (k + 64 < K) {
            load_a(a0, k + 64);                 // prefetch next even chunk
            load_b4(b0, k + 64);
        }
        mma4(a1, b1);                           // consume odd chunk
    }

#pragma unroll
    for (int j = 0; j < 4; ++j) {
        int col = tmg * 64 + j * 16 + l16;
        float bv = bias[col];
        union { v8f v; float f[8]; } c;
        c.v = acc[j];
#pragma unroll
        for (int r = 0; r < 8; ++r) {
            int row = tn * 16 + half * 8 + r;
            float v = c.f[r] + bv;
            if (RELU) v = fmaxf(v, 0.0f);
            out[(size_t)row * M + col] = v;
        }
    }
}

// ---------------------------------------------------------------------------
// Host side: 3 GIN layers.  Inputs (setup_inputs order):
//   0:x 1:src 2:dst 3:w1 4:b1 5:eps1 6:w2 7:b2 8:eps2 9:w3 10:b3 11:eps3
// ---------------------------------------------------------------------------
extern "C" void kernel_launch(void* const* d_in, const int* in_sizes, int n_in,
                              void* d_out, int out_size, void* d_ws, size_t ws_size,
                              hipStream_t stream) {
    (void)n_in; (void)out_size; (void)ws_size;
    const int IN_C = 256, HID = 512, OUT_C = 256;
    const int N = in_sizes[0] / IN_C;       // 50000 (multiple of 16)
    const int E = in_sizes[1];              // 800000

    const float* x   = (const float*)d_in[0];
    const int*   src = (const int*)  d_in[1];
    const int*   dst = (const int*)  d_in[2];
    const float* W[3]   = { (const float*)d_in[3], (const float*)d_in[6], (const float*)d_in[9]  };
    const float* B[3]   = { (const float*)d_in[4], (const float*)d_in[7], (const float*)d_in[10] };
    const float* EPS[3] = { (const float*)d_in[5], (const float*)d_in[8], (const float*)d_in[11] };

    // Workspace carve-up (~257 MB total).
    char* ws = (char*)d_ws;
    float*          agg  = (float*)ws;                                       // N*512 f32
    float*          hbuf = (float*)(ws + (size_t)N * HID * 4);               // N*512 f32
    unsigned short* zbf  = (unsigned short*)(ws + (size_t)N * HID * 8);      // N*512 bf16
    unsigned short* wT   = (unsigned short*)(ws + (size_t)N * HID * 8
                                               + (size_t)N * HID * 2);       // 512*512 bf16

    const int TPB = 256;

    // Shared prologue for a layer: zero agg, scatter, combine->bf16, W->bf16^T.
    auto prologue = [&](const float* hin, int Cin, int Cout,
                        const float* w, const float* eps) {
        int nAgg = N * Cin;
        gin_zero_f32<<<(nAgg + TPB - 1) / TPB, TPB, 0, stream>>>(agg, nAgg);

        int shift = (Cin == 256) ? 6 : 7;               // log2(Cin/4)
        int total = E << shift;
        gin_scatter_add<<<(total + TPB - 1) / TPB, TPB, 0, stream>>>(
            hin, src, dst, agg, Cin, shift, total);

        gin_combine_bf16<<<(nAgg + TPB - 1) / TPB, TPB, 0, stream>>>(
            hin, agg, eps, zbf, nAgg);

        int wtot = Cin * Cout;
        gin_wt_bf16<<<(wtot + TPB - 1) / TPB, TPB, 0, stream>>>(w, wT, Cin, Cout, wtot);
    };

    auto gemm_dims = [&](int Cout, int& nGroupsM, int& blocks) {
        nGroupsM    = Cout / 64;                        // 64-wide column groups
        int nWaves  = (N / 16) * nGroupsM;
        int threads = nWaves * 32;                      // one wave32 per 16x64 tile
        blocks = (threads + TPB - 1) / TPB;
    };

    int nGroupsM, blocks;

    // Layer 1: K=256 -> M=512, ReLU.
    prologue(x, IN_C, HID, W[0], EPS[0]);
    gemm_dims(HID, nGroupsM, blocks);
    gin_gemm_wmma_bf16<256, 512, true><<<blocks, TPB, 0, stream>>>(
        zbf, wT, B[0], hbuf, nGroupsM, (N / 16) * nGroupsM);

    // Layer 2: K=512 -> M=512, ReLU.
    prologue(hbuf, HID, HID, W[1], EPS[1]);
    gemm_dims(HID, nGroupsM, blocks);
    gin_gemm_wmma_bf16<512, 512, true><<<blocks, TPB, 0, stream>>>(
        zbf, wT, B[1], hbuf, nGroupsM, (N / 16) * nGroupsM);

    // Layer 3: K=512 -> M=256, linear.
    prologue(hbuf, HID, OUT_C, W[2], EPS[2]);
    gemm_dims(OUT_C, nGroupsM, blocks);
    gin_gemm_wmma_bf16<512, 256, false><<<blocks, TPB, 0, stream>>>(
        zbf, wT, B[2], (float*)d_out, nGroupsM, (N / 16) * nGroupsM);
}
